// Box2DRoIHead_91173565759783
// MI455X (gfx1250) — compile-verified, matrix operand order checked by reading, not run on hardware
//
#include <hip/hip_runtime.h>
#include <hip/hip_bf16.h>
#include <math.h>

typedef __bf16 bf16;
typedef __attribute__((ext_vector_type(16))) __bf16 v16bf;
typedef __attribute__((ext_vector_type(8)))  __bf16 v8bf;
typedef __attribute__((ext_vector_type(8)))  float  v8f;
typedef __attribute__((ext_vector_type(4)))  int    v4i;

#define GAS __attribute__((address_space(1)))
#define LAS __attribute__((address_space(3)))

// Async global->LDS copy path (gfx1250 ASYNCcnt) if the toolchain exposes it.
#ifndef HAVE_ASYNC_LDS
#if defined(__has_builtin)
#if __has_builtin(__builtin_amdgcn_global_load_async_to_lds_b128)
#define HAVE_ASYNC_LDS 1
#endif
#endif
#endif
#ifndef HAVE_ASYNC_LDS
#define HAVE_ASYNC_LDS 0
#endif

__device__ __forceinline__ void wait_async_zero() {
#if HAVE_ASYNC_LDS
#if defined(__has_builtin) && __has_builtin(__builtin_amdgcn_s_wait_asynccnt)
    __builtin_amdgcn_s_wait_asynccnt(0);
#else
    asm volatile("s_wait_asynccnt 0x0" ::: "memory");
#endif
#else
    // Fallback staging uses ds_store: drain DScnt instead.
#if defined(__has_builtin) && __has_builtin(__builtin_amdgcn_s_wait_dscnt)
    __builtin_amdgcn_s_wait_dscnt(0);
#else
    asm volatile("s_wait_dscnt 0x0" ::: "memory");
#endif
#endif
}

__device__ __forceinline__ void wait_ds_zero() {
#if defined(__has_builtin) && __has_builtin(__builtin_amdgcn_s_wait_dscnt)
    __builtin_amdgcn_s_wait_dscnt(0);
#else
    asm volatile("s_wait_dscnt 0x0" ::: "memory");
#endif
}

// ---------------------------------------------------------------------------
// Weight pre-swizzle: f32 [K,N] row-major -> bf16 WMMA B-fragments.
// Fragment (nt, kt) holds the 32x16 (KxN) tile; storage = frag*512 + lane*16 + e,
// with K index = kt*32 + e + 8*((e>>3) + (lane>>4)), N index = nt*16 + (lane&15),
// matching the CDNA5 16-bit matrix VGPR layout (ISA 7.12.2).
// ---------------------------------------------------------------------------
__global__ void swizzle_w_kernel(const float* __restrict__ W, bf16* __restrict__ out,
                                 int K, int N, long total) {
    long tid = (long)blockIdx.x * blockDim.x + threadIdx.x;
    if (tid >= total) return;
    int  KT   = K >> 5;
    long f    = tid >> 9;
    int  idx  = (int)(tid & 511);
    int  lane = idx >> 4;
    int  e    = idx & 15;
    int  nt   = (int)(f / KT);
    int  kt   = (int)(f % KT);
    int  k    = kt * 32 + e + 8 * ((e >> 3) + (lane >> 4));
    int  n    = nt * 16 + (lane & 15);
    float v   = (n < N) ? W[(size_t)k * N + n] : 0.0f;   // zero-pad N (cls head 81->96)
    out[tid]  = (bf16)v;
}

// ---------------------------------------------------------------------------
// ROI-Align over 4 FPN levels. One block per ROI (1024), one thread per channel
// (256). Bin geometry computed once per block into LDS. Output bf16 row-major
// [roi, c*49 + by*7 + bx]  (== reference transpose(0,3,1,2).reshape).
// Feature maps (~170 MB total) are L2-resident on MI455X (192 MB).
// ---------------------------------------------------------------------------
__global__ __launch_bounds__(256) void roi_align_kernel(
    const float* __restrict__ f0, const float* __restrict__ f1,
    const float* __restrict__ f2, const float* __restrict__ f3,
    const float* __restrict__ boxes, bf16* __restrict__ pooled) {

    const int r = blockIdx.x;     // roi index, b = r/512
    const int c = threadIdx.x;    // channel

    __shared__ float s_x1, s_y1, s_rw, s_rh;
    __shared__ int   s_lvl;
    __shared__ int   sy0[14], sy1[14], sx0[14], sx1[14];
    __shared__ float sly[14], slx[14];
    __shared__ int   syv[14], sxv[14];

    if (c == 0) {
        float x1 = boxes[r * 4 + 0], y1 = boxes[r * 4 + 1];
        float x2 = boxes[r * 4 + 2], y2 = boxes[r * 4 + 3];
        float w = x2 - x1, h = y2 - y1;
        float lf  = floorf(4.0f + log2f(sqrtf(w * h) / 224.0f + 1e-6f));
        int   lvl = (int)fminf(fmaxf(lf, 2.0f), 5.0f) - 2;
        float stride = (float)(4 << lvl);
        s_lvl = lvl;
        float bx1 = x1 / stride, by1 = y1 / stride;
        float bx2 = x2 / stride, by2 = y2 / stride;
        s_x1 = bx1; s_y1 = by1;
        s_rw = fmaxf(bx2 - bx1, 1.0f);
        s_rh = fmaxf(by2 - by1, 1.0f);
    }
    __syncthreads();

    const int lvl = s_lvl;
    const int H   = 256 >> lvl;   // square maps

    if (c < 14) {                                   // y samples
        int   bin = c >> 1, s = c & 1;
        float g   = (float)bin + (s + 0.5f) * 0.5f;
        float yc  = s_y1 + (s_rh / 7.0f) * g;
        syv[c]    = (yc > -1.0f) && (yc < (float)H);
        float cc  = fminf(fmaxf(yc, 0.0f), (float)(H - 1));
        int   i0  = (int)floorf(cc);
        sy0[c] = i0;
        sy1[c] = (i0 + 1 < H - 1) ? (i0 + 1) : (H - 1);
        sly[c] = cc - (float)i0;
    } else if (c >= 32 && c < 46) {                 // x samples
        int   q   = c - 32;
        int   bin = q >> 1, s = q & 1;
        float g   = (float)bin + (s + 0.5f) * 0.5f;
        float xc  = s_x1 + (s_rw / 7.0f) * g;
        sxv[q]    = (xc > -1.0f) && (xc < (float)H);
        float cc  = fminf(fmaxf(xc, 0.0f), (float)(H - 1));
        int   i0  = (int)floorf(cc);
        sx0[q] = i0;
        sx1[q] = (i0 + 1 < H - 1) ? (i0 + 1) : (H - 1);
        slx[q] = cc - (float)i0;
    }
    __syncthreads();

    const float* fmap = (lvl == 0) ? f0 : (lvl == 1) ? f1 : (lvl == 2) ? f2 : f3;
    const int    Wd   = H;
    const int    b    = r >> 9;
    const float* base = fmap + (size_t)(b * 256 + c) * (size_t)H * (size_t)Wd;
    bf16*        outp = pooled + (size_t)r * 12544 + (size_t)c * 49;

    for (int by = 0; by < 7; ++by) {
        for (int bx = 0; bx < 7; ++bx) {
            float acc = 0.0f;
#pragma unroll
            for (int sy = 0; sy < 2; ++sy) {
#pragma unroll
                for (int sx = 0; sx < 2; ++sx) {
                    int qy = by * 2 + sy, qx = bx * 2 + sx;
                    if (syv[qy] && sxv[qx]) {
                        int   y0 = sy0[qy], y1 = sy1[qy];
                        int   x0 = sx0[qx], x1 = sx1[qx];
                        float ly = sly[qy], lx = slx[qx];
                        float v00 = base[(size_t)y0 * Wd + x0];
                        float v01 = base[(size_t)y0 * Wd + x1];
                        float v10 = base[(size_t)y1 * Wd + x0];
                        float v11 = base[(size_t)y1 * Wd + x1];
                        acc += v00 * (1.0f - ly) * (1.0f - lx)
                             + v01 * (1.0f - ly) * lx
                             + v10 * ly * (1.0f - lx)
                             + v11 * ly * lx;
                    }
                }
            }
            outp[by * 7 + bx] = (bf16)(acc * 0.25f);
        }
    }
}

// ---------------------------------------------------------------------------
// WMMA GEMM with LDS-staged, software-pipelined B fragments.
// Block = 4 waves = 128 threads, tile 64(M) x NT*16(N).
// Per K-step of 32: the NT B-fragments (NT KB) are staged into LDS once per
// block (shared by all 4 waves) via async global->LDS copies; the stage for
// step kt+1 is issued before the WMMAs of step kt so the DMA overlaps matrix
// math. Staging is fully unrolled straight-line code with pre-computed chunk
// pointers (no per-lane loop, no exec churn); the final k-step is peeled so
// the in-loop stage is unconditional.
// ---------------------------------------------------------------------------
template <int NT, bool RELU, bool OUT_BF16>
__global__ __launch_bounds__(128) void gemm_wmma_kernel(
    const bf16* __restrict__ A, const bf16* __restrict__ Bp,
    const float* __restrict__ bias, void* __restrict__ outv,
    int K, int Nvalid, int ldOut) {

    const int KT      = K >> 5;
    const int tid     = threadIdx.x;
    const int lane    = tid & 31;
    const int wave    = tid >> 5;
    const int rowTile = blockIdx.x * 64 + wave * 16;
    const int ntBase  = blockIdx.y * NT;
    const int m15     = lane & 15;
    const int hi      = lane >> 4;

    __shared__ __align__(32) bf16 sB[NT * 512];

    // Per-thread staging chunk pointers, hoisted out of the k-loop.
    // NT*64 16-byte chunks over 128 threads -> NCHUNK chunks per thread.
    constexpr int NCHUNK = (NT * 64) / 128 > 0 ? (NT * 64) / 128 : 1;
    const bf16* gsrc[NCHUNK];
#if HAVE_ASYNC_LDS
    LAS v4i*    ldst[NCHUNK];
#else
    bf16*       ldst[NCHUNK];
#endif
#pragma unroll
    for (int u = 0; u < NCHUNK; ++u) {
        int cchunk = tid + u * 128;
        int j      = cchunk >> 6;          // fragment index
        int off    = (cchunk & 63) * 8;    // bf16 element offset in frag
        gsrc[u]    = Bp + (size_t)(ntBase + j) * KT * 512 + off;
#if HAVE_ASYNC_LDS
        ldst[u]    = (LAS v4i*)(sB + j * 512 + off);
#else
        ldst[u]    = sB + j * 512 + off;
#endif
    }

    auto stage = [&](int kt) {
#pragma unroll
        for (int u = 0; u < NCHUNK; ++u) {
#if HAVE_ASYNC_LDS
            __builtin_amdgcn_global_load_async_to_lds_b128(
                (GAS v4i*)(gsrc[u] + (size_t)kt * 512), ldst[u], 0, 0);
#else
            v4i t = *(const v4i*)(gsrc[u] + (size_t)kt * 512);
            *(v4i*)ldst[u] = t;
#endif
        }
    };

    v8f acc[NT] = {};   // zero-init accumulators

    const bf16* arowBase = A + (size_t)(rowTile + m15) * K + 8 * hi;

    stage(0);
    for (int kt = 0; kt < KT - 1; ++kt) {
        wait_async_zero();       // this wave's staged chunks are in LDS
        __syncthreads();         // all waves' chunks are in LDS

        v16bf bfr[NT];
#pragma unroll
        for (int j = 0; j < NT; ++j)
            bfr[j] = *(const v16bf*)(sB + j * 512 + lane * 16);

        const bf16* ap  = arowBase + (size_t)kt * 32;
        v8bf        alo = *(const v8bf*)ap;          // K = kb+0..7
        v8bf        ahi = *(const v8bf*)(ap + 16);   // K = kb+16..23
        __builtin_prefetch(ap + 8 * 32, 0, 3);       // A-stream prefetch, near scope
        v16bf a;
#pragma unroll
        for (int i = 0; i < 8; ++i) { a[i] = alo[i]; a[i + 8] = ahi[i]; }

        wait_ds_zero();          // B fragments landed in registers
        __syncthreads();         // everyone consumed sB; safe to overwrite

        stage(kt + 1);           // unconditional: overlap DMA with WMMAs below

#pragma unroll
        for (int j = 0; j < NT; ++j)
            acc[j] = __builtin_amdgcn_wmma_f32_16x16x32_bf16(
                false, a, false, bfr[j], (short)0, acc[j], false, false);
    }

    {   // peeled final k-step (no next stage)
        const int kt = KT - 1;
        wait_async_zero();
        __syncthreads();

        v16bf bfr[NT];
#pragma unroll
        for (int j = 0; j < NT; ++j)
            bfr[j] = *(const v16bf*)(sB + j * 512 + lane * 16);

        const bf16* ap  = arowBase + (size_t)kt * 32;
        v8bf        alo = *(const v8bf*)ap;
        v8bf        ahi = *(const v8bf*)(ap + 16);
        v16bf a;
#pragma unroll
        for (int i = 0; i < 8; ++i) { a[i] = alo[i]; a[i + 8] = ahi[i]; }

#pragma unroll
        for (int j = 0; j < NT; ++j)
            acc[j] = __builtin_amdgcn_wmma_f32_16x16x32_bf16(
                false, a, false, bfr[j], (short)0, acc[j], false, false);
    }

#pragma unroll
    for (int j = 0; j < NT; ++j) {
        int col = ntBase * 16 + j * 16 + m15;
        if (col < Nvalid) {
            float bv = bias[col];
#pragma unroll
            for (int rr = 0; rr < 8; ++rr) {
                int   row = rowTile + rr + 8 * hi;
                float v   = acc[j][rr] + bv;
                if (RELU) v = v > 0.0f ? v : 0.0f;
                if (OUT_BF16)
                    ((bf16*)outv)[(size_t)row * ldOut + col] = (bf16)v;
                else
                    ((float*)outv)[(size_t)row * ldOut + col] = v;
            }
        }
    }
}

// ---------------------------------------------------------------------------
extern "C" void kernel_launch(void* const* d_in, const int* in_sizes, int n_in,
                              void* d_out, int out_size, void* d_ws, size_t ws_size,
                              hipStream_t stream) {
    const float* f0    = (const float*)d_in[0];
    const float* f1    = (const float*)d_in[1];
    const float* f2    = (const float*)d_in[2];
    const float* f3    = (const float*)d_in[3];
    const float* boxes = (const float*)d_in[4];
    const float* w1    = (const float*)d_in[5];
    const float* b1    = (const float*)d_in[6];
    const float* w2    = (const float*)d_in[7];
    const float* b2    = (const float*)d_in[8];
    const float* wc    = (const float*)d_in[9];
    const float* bc    = (const float*)d_in[10];
    const float* wr    = (const float*)d_in[11];
    const float* br    = (const float*)d_in[12];

    const int M = 1024;                 // B*N rois
    const int DIN = 12544, DFC = 1024;  // 256*7*7, hidden
    const int NC = 81, NCP = 96, NR = 320;

    bf16*  ws = (bf16*)d_ws;
    size_t o  = 0;
    bf16* pooled = ws + o; o += (size_t)M * DIN;
    bf16* w1p    = ws + o; o += (size_t)DIN * DFC;
    bf16* w2p    = ws + o; o += (size_t)DFC * DFC;
    bf16* wcp    = ws + o; o += (size_t)DFC * NCP;
    bf16* wrp    = ws + o; o += (size_t)DFC * NR;
    bf16* h1     = ws + o; o += (size_t)M * DFC;
    bf16* h2     = ws + o; o += (size_t)M * DFC;

    float* cls = (float*)d_out;
    float* reg = (float*)d_out + (size_t)M * NC;

    auto swz = [&](const float* W, bf16* dst, int K, int N, int Npad) {
        long total  = (long)(K >> 5) * (long)(Npad >> 4) * 512;
        long blocks = (total + 255) / 256;
        swizzle_w_kernel<<<(unsigned)blocks, 256, 0, stream>>>(W, dst, K, N, total);
    };
    swz(w1, w1p, DIN, DFC, DFC);
    swz(w2, w2p, DFC, DFC, DFC);
    swz(wc, wcp, DFC, NC,  NCP);
    swz(wr, wrp, DFC, NR,  NR);

    roi_align_kernel<<<M, 256, 0, stream>>>(f0, f1, f2, f3, boxes, pooled);

    dim3 blk(128);
    // FC1: [1024,12544] @ [12544,1024] + b1, ReLU -> bf16 h1
    gemm_wmma_kernel<4, true, true><<<dim3(M / 64, DFC / 64), blk, 0, stream>>>(
        pooled, w1p, b1, h1, DIN, DFC, DFC);
    // FC2: [1024,1024] @ [1024,1024] + b2, ReLU -> bf16 h2
    gemm_wmma_kernel<4, true, true><<<dim3(M / 64, DFC / 64), blk, 0, stream>>>(
        h1, w2p, b2, h2, DFC, DFC, DFC);
    // cls head: N padded 81->96, column-guarded f32 store into d_out
    gemm_wmma_kernel<2, false, false><<<dim3(M / 64, NCP / 32), blk, 0, stream>>>(
        h2, wcp, bc, cls, DFC, NC, NC);
    // bbox head: N = 320
    gemm_wmma_kernel<4, false, false><<<dim3(M / 64, NR / 64), blk, 0, stream>>>(
        h2, wrp, br, reg, DFC, NR, NR);
}
